// CanvasPainterEncoder_70325794505064
// MI455X (gfx1250) — matrix operand order repaired and checked by reading, not compiled, not run on hardware
//
#include <hip/hip_runtime.h>
#include <hip/hip_bf16.h>
#include <math.h>

typedef __bf16 bf16;
typedef __attribute__((ext_vector_type(16))) __bf16 v16bf;
typedef __attribute__((ext_vector_type(8)))  __bf16 v8bf;
typedef __attribute__((ext_vector_type(4)))  __bf16 v4bf;
typedef __attribute__((ext_vector_type(8)))  float  v8f;
typedef __attribute__((ext_vector_type(4)))  unsigned int u32x4;
typedef __attribute__((ext_vector_type(8)))  int i32x8;
typedef __attribute__((ext_vector_type(4)))  int i32x4;

#if __has_builtin(__builtin_amdgcn_tensor_load_to_lds) && \
    __has_builtin(__builtin_amdgcn_s_wait_tensorcnt)
#define HAVE_TDM 1
#else
#define HAVE_TDM 0
#endif

#define NB   64      // batch
#define NTOK 1024    // tokens
#define DM   1024    // d_model = d_state = d_att
#define HH   64
#define WW   64
#define CC   16
#define KK   5
#define PAD  2
#define CW   (WW + 2*PAD)   // 68
#define CH   (HH + 2*PAD)   // 68
#define TSTEPS 32

// ---------------- WMMA fragment helpers (CDNA5 16x16x32 bf16) ----------------

__device__ inline v8f zero_v8f() {
  v8f z;
#pragma unroll
  for (int i = 0; i < 8; ++i) z[i] = 0.0f;
  return z;
}

// A fragment, 16x32 (MxK), source bf16 row-major
__device__ inline v16bf load_a_bf16(const bf16* A, int lda, int row0, int k0) {
  int lane = threadIdx.x & 31;
  int half = lane >> 4;
  const bf16* p = A + (size_t)(row0 + (lane & 15)) * lda + k0 + half * 8;
  v8bf lo = *(const v8bf*)(p);        // K = half*8 .. half*8+7
  v8bf hi = *(const v8bf*)(p + 16);   // K = 16+half*8 ..
  v16bf a;
#pragma unroll
  for (int i = 0; i < 8; ++i) { a[i] = lo[i]; a[8 + i] = hi[i]; }
  return a;
}

// A fragment from fp32 source, converted to bf16 on the fly
__device__ inline v16bf load_a_f32(const float* A, int lda, int row0, int k0) {
  int lane = threadIdx.x & 31;
  int half = lane >> 4;
  const float* p = A + (size_t)(row0 + (lane & 15)) * lda + k0 + half * 8;
  float4 a0 = *(const float4*)(p);
  float4 a1 = *(const float4*)(p + 4);
  float4 a2 = *(const float4*)(p + 16);
  float4 a3 = *(const float4*)(p + 20);
  v16bf a;
  a[0]=(bf16)a0.x; a[1]=(bf16)a0.y; a[2]=(bf16)a0.z; a[3]=(bf16)a0.w;
  a[4]=(bf16)a1.x; a[5]=(bf16)a1.y; a[6]=(bf16)a1.z; a[7]=(bf16)a1.w;
  a[8]=(bf16)a2.x; a[9]=(bf16)a2.y; a[10]=(bf16)a2.z; a[11]=(bf16)a2.w;
  a[12]=(bf16)a3.x; a[13]=(bf16)a3.y; a[14]=(bf16)a3.z; a[15]=(bf16)a3.w;
  return a;
}

// B fragment from pre-packed buffer: fragment (kt, nt16) of a K x Np matrix.
// Packed layout: dst[((nt16*Kt + kt)*32 + lane)*16 + e]  -> 32B contiguous per lane.
__device__ inline v16bf load_b_packed(const bf16* P, int Kt, int kt, int nt16) {
  int lane = threadIdx.x & 31;
  const bf16* p = P + (((size_t)nt16 * Kt + kt) * 32 + lane) * 16;
  v8bf lo = *(const v8bf*)(p);
  v8bf hi = *(const v8bf*)(p + 8);
  v16bf b;
#pragma unroll
  for (int i = 0; i < 8; ++i) { b[i] = lo[i]; b[8 + i] = hi[i]; }
  return b;
}

__device__ inline void store_tile(v8f acc, int row, int col, const float* bias,
                                  float* Cf, bf16* Cb, int ldc, int N, int act) {
  if (col >= N) return;
  float bv = bias ? bias[col] : 0.0f;
#pragma unroll
  for (int r = 0; r < 8; ++r) {
    float v = acc[r] + bv;
    if (act == 1) v = tanhf(v);
    size_t off = (size_t)(row + r) * ldc + col;
    if (Cf) Cf[off] = v;
    if (Cb) Cb[off] = (bf16)v;
  }
}

#define WMMA_BF16(a, b, c) \
  __builtin_amdgcn_wmma_f32_16x16x32_bf16(false, (a), false, (b), (short)0, (c), false, false)

// ---- pack fp32 row-major KxN weight into WMMA B-fragment order (bf16, N padded to 32) ----
__global__ void pack_b_kernel(const float* __restrict__ src, bf16* __restrict__ dst,
                              int K, int N) {
  int Np = ((N + 31) >> 5) << 5;
  int idx = blockIdx.x * 256 + threadIdx.x;
  if (idx >= K * Np) return;
  int e    = idx & 15;
  int lane = (idx >> 4) & 31;
  int rest = idx >> 9;                 // nt16 * Kt + kt
  int Kt = K >> 5;
  int kt = rest % Kt, nt16 = rest / Kt;
  int k   = kt * 32 + (lane >> 4) * 8 + (e & 7) + (e >> 3) * 16;
  int col = nt16 * 16 + (lane & 15);
  dst[idx] = (col < N) ? (bf16)src[(size_t)k * N + col] : (bf16)0.0f;
}

// ---------------- generic bf16 WMMA GEMM: C = act(A@B + bias), B pre-packed ----------------
__global__ __launch_bounds__(256)
void gemm_bf16_kernel(const bf16* __restrict__ A, int lda,
                      const bf16* __restrict__ Pb,
                      const float* __restrict__ bias,
                      float* __restrict__ Cf, bf16* __restrict__ Cb, int ldc,
                      int M, int N, int K, int act) {
  int wid = blockIdx.x * 8 + (threadIdx.x >> 5);
  int ntl = (N + 31) >> 5;
  int mtl = M >> 5;
  if (wid >= mtl * ntl) return;
  int mt = wid / ntl, nt = wid % ntl;
  int r0 = mt * 32, c0 = nt * 32;
  int Kt = K >> 5;
  v8f a00 = zero_v8f(), a01 = zero_v8f(), a10 = zero_v8f(), a11 = zero_v8f();
  for (int kt = 0; kt < Kt; ++kt) {
    v16bf fa0 = load_a_bf16(A, lda, r0, kt * 32);
    v16bf fa1 = load_a_bf16(A, lda, r0 + 16, kt * 32);
    v16bf fb0 = load_b_packed(Pb, Kt, kt, nt * 2);
    v16bf fb1 = load_b_packed(Pb, Kt, kt, nt * 2 + 1);
    a00 = WMMA_BF16(fa0, fb0, a00);
    a01 = WMMA_BF16(fa0, fb1, a01);
    a10 = WMMA_BF16(fa1, fb0, a10);
    a11 = WMMA_BF16(fa1, fb1, a11);
  }
  int lane = threadIdx.x & 31;
  int cn = lane & 15, rb = (lane >> 4) * 8;
  store_tile(a00, r0 + rb,      c0 + cn,      bias, Cf, Cb, ldc, N, act);
  store_tile(a01, r0 + rb,      c0 + 16 + cn, bias, Cf, Cb, ldc, N, act);
  store_tile(a10, r0 + 16 + rb, c0 + cn,      bias, Cf, Cb, ldc, N, act);
  store_tile(a11, r0 + 16 + rb, c0 + 16 + cn, bias, Cf, Cb, ldc, N, act);
}

// ---------------- fused K/V projection: reads token_reps (fp32) once ----------------
__global__ __launch_bounds__(256)
void kv_gemm_kernel(const float* __restrict__ A,
                    const bf16* __restrict__ Pk, const bf16* __restrict__ Pv,
                    const float* __restrict__ bk, const float* __restrict__ bv,
                    bf16* __restrict__ Kout, bf16* __restrict__ Vout) {
  const int NT = DM / 32;                       // 32 col tiles
  const int Kt = DM / 32;
  int wid = blockIdx.x * 8 + (threadIdx.x >> 5);
  int mt = wid / NT, nt = wid % NT;
  int r0 = mt * 32, c0 = nt * 32;
  v8f k00 = zero_v8f(), k01 = zero_v8f(), k10 = zero_v8f(), k11 = zero_v8f();
  v8f v00 = zero_v8f(), v01 = zero_v8f(), v10 = zero_v8f(), v11 = zero_v8f();
  for (int kt = 0; kt < Kt; ++kt) {
    if (kt + 1 < Kt)  // prefetch next A rows (global_prefetch_b8)
      __builtin_prefetch(A + (size_t)(r0 + (threadIdx.x & 31)) * DM + kt * 32 + 32, 0, 1);
    v16bf fa0 = load_a_f32(A, DM, r0, kt * 32);
    v16bf fa1 = load_a_f32(A, DM, r0 + 16, kt * 32);
    v16bf bk0 = load_b_packed(Pk, Kt, kt, nt * 2);
    v16bf bk1 = load_b_packed(Pk, Kt, kt, nt * 2 + 1);
    v16bf bv0 = load_b_packed(Pv, Kt, kt, nt * 2);
    v16bf bv1 = load_b_packed(Pv, Kt, kt, nt * 2 + 1);
    k00 = WMMA_BF16(fa0, bk0, k00);
    k01 = WMMA_BF16(fa0, bk1, k01);
    k10 = WMMA_BF16(fa1, bk0, k10);
    k11 = WMMA_BF16(fa1, bk1, k11);
    v00 = WMMA_BF16(fa0, bv0, v00);
    v01 = WMMA_BF16(fa0, bv1, v01);
    v10 = WMMA_BF16(fa1, bv0, v10);
    v11 = WMMA_BF16(fa1, bv1, v11);
  }
  int lane = threadIdx.x & 31;
  int cn = lane & 15, rb = (lane >> 4) * 8;
  store_tile(k00, r0 + rb,      c0 + cn,      bk, nullptr, Kout, DM, DM, 0);
  store_tile(k01, r0 + rb,      c0 + 16 + cn, bk, nullptr, Kout, DM, DM, 0);
  store_tile(k10, r0 + 16 + rb, c0 + cn,      bk, nullptr, Kout, DM, DM, 0);
  store_tile(k11, r0 + 16 + rb, c0 + 16 + cn, bk, nullptr, Kout, DM, DM, 0);
  store_tile(v00, r0 + rb,      c0 + cn,      bv, nullptr, Vout, DM, DM, 0);
  store_tile(v01, r0 + rb,      c0 + 16 + cn, bv, nullptr, Vout, DM, DM, 0);
  store_tile(v10, r0 + 16 + rb, c0 + cn,      bv, nullptr, Vout, DM, DM, 0);
  store_tile(v11, r0 + 16 + rb, c0 + 16 + cn, bv, nullptr, Vout, DM, DM, 0);
}

// ---------------- elementwise / attention kernels ----------------

// g_bf for s0 GEMM; global_rep copied into x_bf[:,1024:] and li_bf[:,2048:]
__global__ void fillg_kernel(const float* __restrict__ g, bf16* __restrict__ g_bf,
                             bf16* __restrict__ x_bf, bf16* __restrict__ li_bf) {
  int idx = blockIdx.x * 256 + threadIdx.x;   // 64*1024
  int b = idx >> 10, d = idx & 1023;
  bf16 v = (bf16)g[idx];
  g_bf[idx] = v;
  x_bf[(size_t)b * 2048 + 1024 + d] = v;
  li_bf[(size_t)b * 3072 + 2048 + d] = v;
}

// scores[b,n] = dot(q[b,:], k[b,n,:]) + mask  — one wave per (b,n), bandwidth-bound
__global__ __launch_bounds__(256)
void scores_kernel(const bf16* __restrict__ q, const bf16* __restrict__ k,
                   const int* __restrict__ mask, float* __restrict__ scores) {
  int wid = blockIdx.x * 8 + (threadIdx.x >> 5);   // 65536 rows
  int lane = threadIdx.x & 31;
  int b = wid >> 10, n = wid & 1023;
  const bf16* kr = k + ((size_t)b * NTOK + n) * DM;
  const bf16* qr = q + (size_t)b * DM;
  float acc = 0.0f;
#pragma unroll
  for (int it = 0; it < 4; ++it) {
    int off = it * 256 + lane * 8;
    v8bf kv = *(const v8bf*)(kr + off);
    v8bf qv = *(const v8bf*)(qr + off);
#pragma unroll
    for (int e = 0; e < 8; ++e) acc += (float)qv[e] * (float)kv[e];
  }
#pragma unroll
  for (int o = 16; o > 0; o >>= 1) acc += __shfl_xor(acc, o, 32);
  if (lane == 0) scores[wid] = acc + (mask[wid] == 0 ? -1e9f : 0.0f);
}

// in-place softmax over N=1024, one 256-thread block per batch row
__global__ void softmax_kernel(float* __restrict__ scores) {
  __shared__ float red[256];
  int tid = threadIdx.x;
  float* row = scores + (size_t)blockIdx.x * NTOK;
  float v[4];
  float m = -INFINITY;
#pragma unroll
  for (int i = 0; i < 4; ++i) { v[i] = row[tid + i * 256]; m = fmaxf(m, v[i]); }
  red[tid] = m; __syncthreads();
  for (int s = 128; s > 0; s >>= 1) { if (tid < s) red[tid] = fmaxf(red[tid], red[tid + s]); __syncthreads(); }
  m = red[0]; __syncthreads();
  float sum = 0.0f;
#pragma unroll
  for (int i = 0; i < 4; ++i) { v[i] = __expf(v[i] - m); sum += v[i]; }
  red[tid] = sum; __syncthreads();
  for (int s = 128; s > 0; s >>= 1) { if (tid < s) red[tid] += red[tid + s]; __syncthreads(); }
  float inv = 1.0f / red[0];
#pragma unroll
  for (int i = 0; i < 4; ++i) row[tid + i * 256] = v[i] * inv;
}

// partial r[b,d] over a 256-token slice: 4 consecutive d per thread (8B loads,
// 256B contiguous per wave per token); fp32 atomic accumulation into r32.
// Alpha slice (1 KB) staged into LDS via the Tensor Data Mover when available.
__global__ void rvec_partial_kernel(const float* __restrict__ alpha, const bf16* __restrict__ v,
                                    float* __restrict__ r32) {
  __shared__ float sa[256];
  int b = blockIdx.x >> 2;
  int nbase = (blockIdx.x & 3) << 8;
#if HAVE_TDM
  if ((threadIdx.x >> 5) == 0) {   // wave 0 issues one DMA for the whole block
    unsigned lds_addr = (unsigned)(unsigned long long)
        (__attribute__((address_space(3))) float*)sa;
    unsigned long long ga =
        (unsigned long long)(const void*)(alpha + (size_t)b * NTOK + nbase);
    u32x4 g0;
    g0[0] = 1u;                                           // count=1, user mode
    g0[1] = lds_addr;                                     // LDS byte address
    g0[2] = (unsigned)ga;                                 // global addr [31:0]
    g0[3] = (unsigned)((ga >> 32) & 0x1FFFFFFu) | (2u << 30);  // addr[56:32] | type=2
    i32x8 g1;
    g1[0] = (int)(2u << 16);    // data_size = 4 bytes
    g1[1] = (int)(256u << 16);  // tensor_dim0 = 256 (bits 79:48, low half)
    g1[2] = (int)(1u << 16);    // tensor_dim1 = 1
    g1[3] = (int)(256u << 16);  // tile_dim0 = 256
    g1[4] = 1;                  // tile_dim1 = 1
    g1[5] = 256;                // tensor_dim0_stride = 256
    g1[6] = 0;
    g1[7] = 0;
    i32x4 z4;
    z4[0] = z4[1] = z4[2] = z4[3] = 0;
#if defined(__clang_major__) && __clang_major__ >= 23
    i32x8 z8;
#pragma unroll
    for (int i = 0; i < 8; ++i) z8[i] = 0;
    __builtin_amdgcn_tensor_load_to_lds(g0, g1, z4, z4, z8, 0);
#else
    __builtin_amdgcn_tensor_load_to_lds(g0, g1, z4, z4, 0);
#endif
    __builtin_amdgcn_s_wait_tensorcnt(0);
  }
  __syncthreads();
#else
  sa[threadIdx.x] = alpha[(size_t)b * NTOK + nbase + threadIdx.x];
  __syncthreads();
#endif
  int d0 = threadIdx.x * 4;
  float a0 = 0.0f, a1 = 0.0f, a2 = 0.0f, a3 = 0.0f;
  const bf16* vp = v + ((size_t)b * NTOK + nbase) * DM + d0;
#pragma unroll 4
  for (int n = 0; n < 256; ++n) {
    v4bf vv = *(const v4bf*)(vp + (size_t)n * DM);
    float an = sa[n];
    a0 += an * (float)vv[0];
    a1 += an * (float)vv[1];
    a2 += an * (float)vv[2];
    a3 += an * (float)vv[3];
  }
  float* rp = r32 + (size_t)b * DM + d0;
  atomicAdd(rp + 0, a0);
  atomicAdd(rp + 1, a1);
  atomicAdd(rp + 2, a2);
  atomicAdd(rp + 3, a3);
}

// finalize r: bf16 into x_bf[:, :1024] and li_bf[:, 1024:2048]; re-zero r32 for next step
__global__ void rfin_kernel(float* __restrict__ r32, bf16* __restrict__ x_bf,
                            bf16* __restrict__ li_bf) {
  int idx = blockIdx.x * 256 + threadIdx.x;   // 64*1024
  int b = idx >> 10, d = idx & 1023;
  float val = r32[idx];
  r32[idx] = 0.0f;
  x_bf[(size_t)b * 2048 + d] = (bf16)val;
  li_bf[(size_t)b * 3072 + 1024 + d] = (bf16)val;
}

// GRU gate fusion: s_new from gi, gh, s; writes s (f32), s_bf, li_bf[:, :1024]
__global__ void gates_kernel(const float* __restrict__ gi, const float* __restrict__ gh,
                             float* __restrict__ s, bf16* __restrict__ s_bf,
                             bf16* __restrict__ li_bf) {
  int idx = blockIdx.x * 256 + threadIdx.x;   // 64*1024
  int b = idx >> 10, d = idx & 1023;
  size_t g0 = (size_t)b * 3072 + d;
  float ir = gi[g0], iz = gi[g0 + 1024], in_ = gi[g0 + 2048];
  float hr = gh[g0], hz = gh[g0 + 1024], hn  = gh[g0 + 2048];
  float rg = 1.0f / (1.0f + __expf(-(ir + hr)));
  float zg = 1.0f / (1.0f + __expf(-(iz + hz)));
  float ng = tanhf(in_ + rg * hn);
  float snew = (1.0f - zg) * ng + zg * s[idx];
  s[idx] = snew;
  s_bf[idx] = (bf16)snew;
  li_bf[g0] = (bf16)snew;
}

// location head: li[b,:3072] @ W_loc[:, 0:2] -> (cx, cy) -> grid (i, j)
__global__ void loc_kernel(const bf16* __restrict__ li, const float* __restrict__ Wloc,
                           const float* __restrict__ bloc, int* __restrict__ ij) {
  int b = blockIdx.x, lane = threadIdx.x;   // 32 threads
  float a0 = 0.0f, a1 = 0.0f;
  for (int e = lane; e < 3072; e += 32) {
    float x = (float)li[(size_t)b * 3072 + e];
    a0 += x * Wloc[e * 3 + 0];
    a1 += x * Wloc[e * 3 + 1];
  }
#pragma unroll
  for (int o = 16; o > 0; o >>= 1) { a0 += __shfl_xor(a0, o, 32); a1 += __shfl_xor(a1, o, 32); }
  if (lane == 0) {
    float cx = tanhf(a0 + bloc[0]);
    float cy = tanhf(a1 + bloc[1]);
    int j = min(WW - 1, max(0, (int)rintf(0.5f * (cx + 1.0f) * (WW - 1))));
    int i = min(HH - 1, max(0, (int)rintf(0.5f * (cy + 1.0f) * (HH - 1))));
    ij[b * 2 + 0] = i;
    ij[b * 2 + 1] = j;
  }
}

// scatter-add patch[b, c*25+ky*5+kx] into padded canvas [B, 68, 68, 16]
__global__ void paint_kernel(const float* __restrict__ patch, const int* __restrict__ ij,
                             float* __restrict__ canvas) {
  int b = blockIdx.x, t = threadIdx.x;
  if (t >= CC * KK * KK) return;
  int c = t / (KK * KK), rem = t % (KK * KK), ky = rem / KK, kx = rem % KK;
  int i = ij[b * 2], j = ij[b * 2 + 1];
  size_t off = (((size_t)b * CH + (i + ky)) * CW + (j + kx)) * CC + c;
  canvas[off] += patch[(size_t)b * (CC * KK * KK) + t];
}

// crop + BHWC -> BCHW
__global__ void crop_kernel(const float* __restrict__ canvas, float* __restrict__ out) {
  int idx = blockIdx.x * 256 + threadIdx.x;   // 64*16*64*64
  int w = idx & 63, h = (idx >> 6) & 63, c = (idx >> 12) & 15, b = idx >> 16;
  out[idx] = canvas[(((size_t)b * CH + (h + PAD)) * CW + (w + PAD)) * CC + c];
}

// ---------------- host ----------------

extern "C" void kernel_launch(void* const* d_in, const int* in_sizes, int n_in,
                              void* d_out, int out_size, void* d_ws, size_t ws_size,
                              hipStream_t stream) {
  (void)in_sizes; (void)n_in; (void)out_size; (void)ws_size;
  const float* tok    = (const float*)d_in[0];
  const float* grep   = (const float*)d_in[1];
  const int*   mask   = (const int*)  d_in[2];
  const float* W_init = (const float*)d_in[3];
  const float* b_init = (const float*)d_in[4];
  const float* Wq     = (const float*)d_in[5];
  const float* bq     = (const float*)d_in[6];
  const float* Wk     = (const float*)d_in[7];
  const float* bk     = (const float*)d_in[8];
  const float* Wv     = (const float*)d_in[9];
  const float* bv     = (const float*)d_in[10];
  const float* W_ih   = (const float*)d_in[11];
  const float* b_ih   = (const float*)d_in[12];
  const float* W_hh   = (const float*)d_in[13];
  const float* b_hh   = (const float*)d_in[14];
  const float* W_wr   = (const float*)d_in[15];
  const float* b_wr   = (const float*)d_in[16];
  const float* W_loc  = (const float*)d_in[17];
  const float* b_loc  = (const float*)d_in[18];
  float* out = (float*)d_out;

  // workspace carve-up (256B aligned)
  char* ws = (char*)d_ws;
  size_t off = 0;
  auto carve = [&](size_t bytes) -> char* {
    char* p = ws + off;
    off += (bytes + 255) & ~(size_t)255;
    return p;
  };
  bf16* k_bf    = (bf16*)carve((size_t)NB * NTOK * DM * 2);   // 128 MB
  bf16* v_bf    = (bf16*)carve((size_t)NB * NTOK * DM * 2);   // 128 MB
  bf16* Pq      = (bf16*)carve((size_t)DM * DM * 2);          // packed weights
  bf16* Pk      = (bf16*)carve((size_t)DM * DM * 2);
  bf16* Pv      = (bf16*)carve((size_t)DM * DM * 2);
  bf16* Pih     = (bf16*)carve((size_t)2048 * 3072 * 2);
  bf16* Phh     = (bf16*)carve((size_t)1024 * 3072 * 2);
  bf16* Pwr     = (bf16*)carve((size_t)2048 * 416 * 2);       // 400 padded to 416
  bf16* Pini    = (bf16*)carve((size_t)DM * DM * 2);
  bf16* g_bf    = (bf16*)carve((size_t)NB * DM * 2);
  float* s      = (float*)carve((size_t)NB * DM * 4);
  bf16* s_bf    = (bf16*)carve((size_t)NB * DM * 2);
  bf16* q_bf    = (bf16*)carve((size_t)NB * DM * 2);
  float* scores = (float*)carve((size_t)NB * NTOK * 4);
  float* r32    = (float*)carve((size_t)NB * DM * 4);
  bf16* x_bf    = (bf16*)carve((size_t)NB * 2048 * 2);
  bf16* li_bf   = (bf16*)carve((size_t)NB * 3072 * 2);
  float* gi     = (float*)carve((size_t)NB * 3072 * 4);
  float* gh     = (float*)carve((size_t)NB * 3072 * 4);
  float* patch  = (float*)carve((size_t)NB * 400 * 4);
  int*   ij     = (int*)  carve((size_t)NB * 2 * 4);
  float* canvas = (float*)carve((size_t)NB * CH * CW * CC * 4);

  dim3 blk(256);
  auto pack = [&](const float* src, bf16* dst, int K, int N) {
    int Np = ((N + 31) / 32) * 32;
    pack_b_kernel<<<(K * Np + 255) / 256, blk, 0, stream>>>(src, dst, K, N);
  };
  // phase 0: pack weights (fused f32->bf16 cast), g fill, zeroing, K/V projection, s0
  pack(Wq, Pq, 1024, 1024);
  pack(Wk, Pk, 1024, 1024);
  pack(Wv, Pv, 1024, 1024);
  pack(W_ih, Pih, 2048, 3072);
  pack(W_hh, Phh, 1024, 3072);
  pack(W_wr, Pwr, 2048, 400);
  pack(W_init, Pini, 1024, 1024);
  fillg_kernel<<<(NB * DM) / 256, blk, 0, stream>>>(grep, g_bf, x_bf, li_bf);
  hipMemsetAsync(canvas, 0, (size_t)NB * CH * CW * CC * 4, stream);
  hipMemsetAsync(r32, 0, (size_t)NB * DM * 4, stream);

  // K/V: M = 65536, N = 1024 -> 65536 waves / 8 per block
  kv_gemm_kernel<<<(NB * NTOK / 32) * (DM / 32) / 8, blk, 0, stream>>>(
      tok, Pk, Pv, bk, bv, k_bf, v_bf);

  auto gemm = [&](const bf16* A, int lda, const bf16* Pb, const float* bias,
                  float* Cf, bf16* Cb, int ldc, int M, int N, int K, int act) {
    int waves = (M / 32) * ((N + 31) / 32);
    gemm_bf16_kernel<<<(waves + 7) / 8, blk, 0, stream>>>(A, lda, Pb, bias, Cf, Cb, ldc,
                                                          M, N, K, act);
  };
  // s0 = tanh(g @ W_init + b_init)
  gemm(g_bf, DM, Pini, b_init, s, s_bf, DM, NB, DM, DM, 1);

  for (int t = 0; t < TSTEPS; ++t) {
    // q = s @ Wq + bq  (bf16 out)
    gemm(s_bf, DM, Pq, bq, nullptr, q_bf, DM, NB, DM, DM, 0);
    // attention scores + mask, softmax, r = alpha @ v
    scores_kernel<<<(NB * NTOK) / 8, blk, 0, stream>>>(q_bf, k_bf, mask, scores);
    softmax_kernel<<<NB, blk, 0, stream>>>(scores);
    rvec_partial_kernel<<<NB * 4, blk, 0, stream>>>(scores, v_bf, r32);
    rfin_kernel<<<(NB * DM) / 256, blk, 0, stream>>>(r32, x_bf, li_bf);
    // GRU GEMMs
    gemm(x_bf, 2048, Pih, b_ih, gi, nullptr, 3072, NB, 3072, 2048, 0);
    gemm(s_bf, 1024, Phh, b_hh, gh, nullptr, 3072, NB, 3072, 1024, 0);
    gates_kernel<<<(NB * DM) / 256, blk, 0, stream>>>(gi, gh, s, s_bf, li_bf);
    // patch = [s_new, r] @ W_write + b_write   (wi == li[:, :2048])
    gemm(li_bf, 3072, Pwr, b_wr, patch, nullptr, 400, NB, 400, 2048, 0);
    // location + paint
    loc_kernel<<<NB, dim3(32), 0, stream>>>(li_bf, W_loc, b_loc, ij);
    paint_kernel<<<NB, dim3(416), 0, stream>>>(patch, ij, canvas);
  }
  crop_kernel<<<(NB * CC * HH * WW) / 256, blk, 0, stream>>>(canvas, out);
}
